// BarrierNet_19610820673836
// MI455X (gfx1250) — matrix operand order repaired and checked by reading, not compiled, not run on hardware
//
#include <hip/hip_runtime.h>

typedef _Float16 h8  __attribute__((ext_vector_type(8)));
typedef _Float16 h16 __attribute__((ext_vector_type(16)));
typedef float    f8  __attribute__((ext_vector_type(8)));

#if __has_builtin(__builtin_amdgcn_sched_barrier)
#define SCHED_FENCE() __builtin_amdgcn_sched_barrier(0)
#else
#define SCHED_FENCE()
#endif

static __device__ __forceinline__ h16 cat8(h8 a, h8 b) {
    return __builtin_shufflevector(a, b, 0,1,2,3,4,5,6,7,8,9,10,11,12,13,14,15);
}

// exchange a v8h with the partner lane in the other 16-lane half (lane ^ 16)
static __device__ __forceinline__ h8 xhalf(h8 v) {
    union { h8 h; int i[4]; } u;
    u.h = v;
#pragma unroll
    for (int k = 0; k < 4; ++k) u.i[k] = __shfl_xor(u.i[k], 16, 32);
    return u.h;
}

static __device__ __forceinline__ f8 wmma_f16(h16 a, h16 b, f8 c) {
    return __builtin_amdgcn_wmma_f32_16x16x32_f16(
        /*neg_a=*/false, a, /*neg_b=*/false, b,
        /*c_mod=*/(short)0, c, /*reuse_a=*/false, /*reuse_b=*/false);
}

// A-operand chunk: row-major f16 weight row, K-chunk c (32 wide).
// element e <-> K = 32c + 16*hi + e  => two contiguous v8h loads
static __device__ __forceinline__ h16 loadA(const _Float16* W, int row, int stride,
                                            int c, int hi) {
    const _Float16* p = W + row * stride + 32 * c + 16 * hi;
    h8 lo = *reinterpret_cast<const h8*>(p);
    h8 hh = *reinterpret_cast<const h8*>(p + 8);
    return cat8(lo, hh);
}

#define OBS_X 4.0f
#define OBS_Y 6.0f
#define RAD2  2.25f

__global__ void __launch_bounds__(256)
barriernet_wmma_kernel(const float* __restrict__ x,
                       const float* __restrict__ mean, const float* __restrict__ stdv,
                       const float* __restrict__ W1,  const float* __restrict__ b1,
                       const float* __restrict__ W21, const float* __restrict__ b21,
                       const float* __restrict__ W22, const float* __restrict__ b22,
                       const float* __restrict__ W31, const float* __restrict__ b31,
                       const float* __restrict__ W32, const float* __restrict__ b32,
                       float* __restrict__ out, int n)
{
    // zero-padded f16 weights: every lane can load its A-chunk unconditionally
    __shared__ __align__(16) _Float16 sW1p[128 * 32];   // K padded 8 -> 32
    __shared__ __align__(16) _Float16 sW21[64 * 128];
    __shared__ __align__(16) _Float16 sW22[64 * 128];
    __shared__ __align__(16) _Float16 sW3p[16 * 64];    // rows 0-1 = W31, rest 0
    __shared__ __align__(16) _Float16 sW3q[16 * 64];    // rows 0-1 = W32, rest 0
    __shared__ float sB1[128], sB21[64], sB22[64], sB3[4];

    const int tid = threadIdx.x;
    for (int i = tid; i < 128 * 32; i += 256) {
        int m = i >> 5, k = i & 31;
        sW1p[i] = (k < 8) ? (_Float16)W1[m * 8 + k] : (_Float16)0.0f;
    }
    for (int i = tid; i < 64 * 128; i += 256) {
        sW21[i] = (_Float16)W21[i];
        sW22[i] = (_Float16)W22[i];
    }
    for (int i = tid; i < 16 * 64; i += 256) {
        int r = i >> 6, k = i & 63;
        sW3p[i] = (r < 2) ? (_Float16)W31[r * 64 + k] : (_Float16)0.0f;
        sW3q[i] = (r < 2) ? (_Float16)W32[r * 64 + k] : (_Float16)0.0f;
    }
    if (tid < 128) sB1[tid] = b1[tid];
    if (tid < 64) { sB21[tid] = b21[tid]; sB22[tid] = b22[tid]; }
    if (tid < 4)  sB3[tid] = (tid < 2) ? b31[tid] : b32[tid - 2];
    __syncthreads();

    const int lane = tid & 31;
    const int ln   = lane & 15;     // sample column / A-row within tile
    const int hi   = lane >> 4;     // lane half
    const int wv   = tid >> 5;

    const int nwaves = gridDim.x * 8;
    const int wid    = blockIdx.x * 8 + wv;
    const int ntiles = n >> 4;

    float mn[8], sd[8];
#pragma unroll
    for (int i = 0; i < 8; ++i) { mn[i] = mean[i]; sd[i] = stdv[i]; }

    for (int t = wid; t < ntiles; t += nwaves) {
        const int s = t * 16 + ln;

        // ---- load x (all lanes; both halves fetch the same row, coalesced),
        //      normalize, build layer-1 B operand (K=8 real, rest zero)
        const float4* xp = reinterpret_cast<const float4*>(x + (size_t)s * 8);
        float4 xa = xp[0], xb = xp[1];
        float x0[8] = {xa.x, xa.y, xa.z, xa.w, xb.x, xb.y, xb.z, xb.w};
#pragma unroll
        for (int i = 0; i < 8; ++i) x0[i] = x0[i] * sd[i] + mn[i];
        const float px = x0[0], py = x0[1], th = x0[2], vv = x0[3];

        h16 B1 = {};
#pragma unroll
        for (int i = 0; i < 8; ++i)
            B1[i] = hi ? (_Float16)0.0f : (_Float16)x0[i];   // K = 16*hi + i
        SCHED_FENCE();

        // ---- layer 1 fused with B2 packing: per chunk c, produce h m-tiles
        //      2c and 2c+1, relu+f16-pack, cross-half exchange -> B2[c]
        h16 B2[4];
#pragma unroll
        for (int c = 0; c < 4; ++c) {
            h8 ha, hb;
            {
                h16 a = loadA(sW1p, 16 * (2 * c) + ln, 32, 0, hi);
                f8 acc = {};
                acc = wmma_f16(a, B1, acc);
#pragma unroll
                for (int r = 0; r < 8; ++r)
                    ha[r] = (_Float16)fmaxf(acc[r] + sB1[32 * c + r + 8 * hi], 0.0f);
            }
            {
                h16 a = loadA(sW1p, 16 * (2 * c + 1) + ln, 32, 0, hi);
                f8 acc = {};
                acc = wmma_f16(a, B1, acc);
#pragma unroll
                for (int r = 0; r < 8; ++r)
                    hb[r] = (_Float16)fmaxf(acc[r] + sB1[32 * c + 16 + r + 8 * hi], 0.0f);
            }
            h8 oa = xhalf(ha);
            h8 ob = xhalf(hb);
            B2[c] = hi ? cat8(ob, hb) : cat8(ha, oa);
            SCHED_FENCE();
        }

        // ---- layers 21 & 22: x21.T = relu(W21 h.T + b21), x22.T likewise
        h8 x21h[4], x22h[4];
#pragma unroll
        for (int mt = 0; mt < 4; ++mt) {
            f8 a21 = {}, a22 = {};
#pragma unroll
            for (int c = 0; c < 4; ++c) {
                h16 w21 = loadA(sW21, 16 * mt + ln, 128, c, hi);
                a21 = wmma_f16(w21, B2[c], a21);
                h16 w22 = loadA(sW22, 16 * mt + ln, 128, c, hi);
                a22 = wmma_f16(w22, B2[c], a22);
            }
#pragma unroll
            for (int r = 0; r < 8; ++r) {
                x21h[mt][r] = (_Float16)fmaxf(a21[r] + sB21[16 * mt + r + 8 * hi], 0.0f);
                x22h[mt][r] = (_Float16)fmaxf(a22[r] + sB22[16 * mt + r + 8 * hi], 0.0f);
            }
            SCHED_FENCE();
        }

        // ---- layer 3: p = W31 x21.T, z = W32 x22.T (rows 0..1 of padded A)
        f8 accp = {}, accq = {};
#pragma unroll
        for (int c = 0; c < 2; ++c) {
            h8 oa = xhalf(x21h[2 * c]);
            h8 ob = xhalf(x21h[2 * c + 1]);
            h8 oc = xhalf(x22h[2 * c]);
            h8 od = xhalf(x22h[2 * c + 1]);
            h16 Bp = hi ? cat8(ob, x21h[2 * c + 1]) : cat8(x21h[2 * c], oa);
            h16 Bq = hi ? cat8(od, x22h[2 * c + 1]) : cat8(x22h[2 * c], oc);

            h16 Ap = loadA(sW3p, ln, 64, c, hi);   // rows >= 2 are zero-padded
            h16 Aq = loadA(sW3q, ln, 64, c, hi);
            accp = wmma_f16(Ap, Bp, accp);
            accq = wmma_f16(Aq, Bq, accq);
            SCHED_FENCE();
        }

        // ---- CBF-QP elementwise tail + store (lanes 0..15 hold M=0,1 results)
        if (hi == 0) {
            float p0 = accp[0] + sB3[0];
            float p1 = accp[1] + sB3[1];
            float z0 = accq[0] + sB3[2];
            float z1 = accq[1] + sB3[3];
            float pen0 = 4.0f / (1.0f + __expf(-z0));
            float pen1 = 4.0f / (1.0f + __expf(-z1));

            float dx = px - OBS_X, dy = py - OBS_Y;
            float barrier = dx * dx + dy * dy - RAD2;
            float st = __sinf(th), ct = __cosf(th);
            float bdot = 2.0f * dx * vv * ct + 2.0f * dy * vv * st;
            float Lf2b = 2.0f * vv * vv;
            float L1 = -2.0f * dx * vv * st + 2.0f * dy * vv * ct;
            float L2 = 2.0f * dx * ct + 2.0f * dy * st;
            float G0 = -L1, G1 = -L2;
            float hcon = Lf2b + (pen0 + pen1) * bdot + pen0 * pen1 * barrier;
            float Gp = G0 * p0 + G1 * p1;
            float GG = G0 * G0 + G1 * G1;
            float lam = fmaxf((-Gp - hcon) / (GG + 1e-12f), 0.0f);

            float2 u;
            u.x = -p0 - lam * G0;
            u.y = -p1 - lam * G1;
            reinterpret_cast<float2*>(out)[s] = u;
        }
    }
}

extern "C" void kernel_launch(void* const* d_in, const int* in_sizes, int n_in,
                              void* d_out, int out_size, void* d_ws, size_t ws_size,
                              hipStream_t stream) {
    const float* x    = (const float*)d_in[0];
    const float* mean = (const float*)d_in[1];
    const float* stdv = (const float*)d_in[2];
    const float* W1   = (const float*)d_in[3];
    const float* b1   = (const float*)d_in[4];
    const float* W21  = (const float*)d_in[5];
    const float* b21  = (const float*)d_in[6];
    const float* W22  = (const float*)d_in[7];
    const float* b22  = (const float*)d_in[8];
    const float* W31  = (const float*)d_in[9];
    const float* b31  = (const float*)d_in[10];
    const float* W32  = (const float*)d_in[11];
    const float* b32  = (const float*)d_in[12];
    float* out = (float*)d_out;

    const int n = in_sizes[0] / 8;              // B samples
    const int ntiles = n / 16;
    int blocks = (ntiles + 7) / 8;
    if (blocks > 1024) blocks = 1024;           // ~2 tiles per wave, amortize LDS fill
    if (blocks < 1) blocks = 1;

    barriernet_wmma_kernel<<<blocks, 256, 0, stream>>>(
        x, mean, stdv, W1, b1, W21, b21, W22, b22, W31, b31, W32, b32, out, n);
}
